// GraphAutoEncoder_89275190215576
// MI455X (gfx1250) — matrix operand order repaired and checked by reading, not compiled, not run on hardware
//
#include <hip/hip_runtime.h>
#include <hip/hip_bf16.h>

typedef __attribute__((ext_vector_type(16))) _Float16 v16h;
typedef __attribute__((ext_vector_type(8)))  _Float16 v8h;
typedef __attribute__((ext_vector_type(8)))  float    v8f;

#define BATCH 64
#define NPTS  128
#define HID   64

// d_out layout (floats), concatenated in reference return order
#define OUT0_OFF    0         // batch[:,:,:4]   64*128*4
#define OUT1_OFF    32768     // batch[:,:,4]    64*128
#define LOGITS_OFF  40960     // logits          64*128*4
#define VALUES_OFF  73728     // values          64*128
#define LATENT_OFF  81920     // latent          64*128*3
#define MASKS_OFF   106496    // masks [3,B,N,N] 3*64*128*128

// workspace layout (floats)
#define WS_SKIP 0
#define WS_X1   524288
#define WS_X2   1048576
#define WS_X3   1572864
#define WS_X4   2097152
#define WS_ADJF 2621440   // adj uint8 [B,N,N] starts at this float offset

// dynamic LDS partition (floats) for gat_kernel
#define LDS_XL    0                    // f32 [128][64]
#define LDS_XR    (NPTS * HID)         // f32 [128][64], overlapped by alpha f16
#define LDS_ATT   (2 * NPTS * HID)     // f32 [64]
#define LDS_XLT   (2 * NPTS * HID + 64) // f16 [64][128] => 4096 floats
#define LDS_TOTAL (2 * NPTS * HID + 64 + NPTS * HID / 2)

// ---------------------------------------------------------------------------
// Encoder MLP: 5 -> 64 -> 64 -> 3, plus skip = 0.1*(latent @ skip_w + skip_b)
// One 64-thread block per (b,n) row.
// ---------------------------------------------------------------------------
__global__ __launch_bounds__(64) void encoder_kernel(
    const float* __restrict__ batch,
    const float* __restrict__ ew1, const float* __restrict__ eb1,
    const float* __restrict__ ew2, const float* __restrict__ eb2,
    const float* __restrict__ ew3, const float* __restrict__ eb3,
    const float* __restrict__ skip_w, const float* __restrict__ skip_b,
    float* __restrict__ outf, float* __restrict__ skip)
{
    __shared__ float h1[64], h2[64], lat[3];
    const int bn = blockIdx.x;
    const int t  = threadIdx.x;
    const float* xrow = batch + (size_t)bn * 5;

    float acc = eb1[t];
    #pragma unroll
    for (int c = 0; c < 5; ++c) acc += xrow[c] * ew1[c * 64 + t];
    h1[t] = fmaxf(acc, 0.0f);
    __syncthreads();

    acc = eb2[t];
    #pragma unroll 8
    for (int k = 0; k < 64; ++k) acc += h1[k] * ew2[k * 64 + t];
    h2[t] = fmaxf(acc, 0.0f);
    __syncthreads();

    if (t < 3) {
        float v = eb3[t];
        #pragma unroll 8
        for (int k = 0; k < 64; ++k) v += h2[k] * ew3[k * 3 + t];
        lat[t] = v;
        outf[LATENT_OFF + (size_t)bn * 3 + t] = v;
    }
    if (t < 4) outf[OUT0_OFF + (size_t)bn * 4 + t] = xrow[t];
    if (t == 4) outf[OUT1_OFF + bn] = xrow[4];
    __syncthreads();

    float s = skip_b[t];
    #pragma unroll
    for (int c = 0; c < 3; ++c) s += lat[c] * skip_w[c * 64 + t];
    skip[WS_SKIP + (size_t)bn * 64 + t] = 0.1f * s;
}

// ---------------------------------------------------------------------------
// Beta-skeleton masks (beta = 0,1,2). One block per batch sample.
// Strict '<' tests done in squared distances (sqrt-free, monotone-equivalent).
// Writes masks [3,B,N,N] as 0/1 floats and adjacency (beta=2 | eye) as u8.
// ---------------------------------------------------------------------------
__global__ __launch_bounds__(256) void mask_kernel(
    const float* __restrict__ latent,   // [B,N,3] (d_out latent region)
    float* __restrict__ outf,
    unsigned char* __restrict__ adj)
{
    __shared__ float pS[NPTS * 3];
    const int b = blockIdx.x;
    const int tid = threadIdx.x;
    for (int idx = tid; idx < NPTS * 3; idx += 256)
        pS[idx] = latent[(size_t)b * NPTS * 3 + idx];
    __syncthreads();

    float* mout = outf + MASKS_OFF;
    const size_t mstride = (size_t)BATCH * NPTS * NPTS;  // between beta planes

    for (int pair = tid; pair < NPTS * NPTS; pair += 256) {
        const int i = pair >> 7;
        const int j = pair & 127;
        float m0, m1, m2;
        unsigned char a;
        if (i == j) {
            m0 = m1 = m2 = 0.0f;
            a = 1;  // self loop in adj
        } else {
            const float pix = pS[i*3+0], piy = pS[i*3+1], piz = pS[i*3+2];
            const float pjx = pS[j*3+0], pjy = pS[j*3+1], pjz = pS[j*3+2];
            const float dx = pix - pjx, dy = piy - pjy, dz = piz - pjz;
            const float D2 = dx*dx + dy*dy + dz*dz;
            const float mx = (pix + pjx) * 0.5f;
            const float my = (piy + pjy) * 0.5f;
            const float mz = (piz + pjz) * 0.5f;
            const float t1 = 0.25f * D2;
            const float t2 = 0.0625f * D2;
            bool k0 = false, k1 = false, k2 = false;
            for (int k = 0; k < NPTS; ++k) {
                if (k == i || k == j) continue;
                const float kx = pS[k*3+0], ky = pS[k*3+1], kz = pS[k*3+2];
                float ax = kx - pix, ay = ky - piy, az = kz - piz;
                const float dki2 = ax*ax + ay*ay + az*az;
                ax = kx - pjx; ay = ky - pjy; az = kz - pjz;
                const float dkj2 = ax*ax + ay*ay + az*az;
                ax = kx - mx; ay = ky - my; az = kz - mz;
                const float dk2 = ax*ax + ay*ay + az*az;
                k0 = k0 || (fmaxf(dki2, dkj2) < D2);
                k1 = k1 || (dk2 < t1);
                k2 = k2 || (dk2 < t2);
                if (k0 && k1 && k2) break;
            }
            m0 = k0 ? 0.0f : 1.0f;
            m1 = k1 ? 0.0f : 1.0f;
            m2 = k2 ? 0.0f : 1.0f;
            a = k2 ? 0 : 1;
        }
        const size_t base = (size_t)b * NPTS * NPTS + (size_t)i * NPTS + j;
        mout[base]               = m0;
        mout[base + mstride]     = m1;
        mout[base + 2 * mstride] = m2;
        adj[base] = a;
    }
}

// ---------------------------------------------------------------------------
// Fused GATv2 layer (heads=1). One block (8 waves) per batch sample.
// LDS: xl f32 [128][64] | xr f32 [128][64] (overlapped by alpha f16 [128][128])
//      | att f32 [64] | xlT f16 [64][128] (pre-converted, transposed).
// Aggregation alpha @ xl via v_wmma_f32_16x16x32_f16 with b128 LDS fragment
// loads (no converts, no strided reads in the MMA loop).
// ---------------------------------------------------------------------------
__global__ __launch_bounds__(256) void gat_kernel(
    const float* __restrict__ x, int xstride, int xoff, int cin,
    const float* __restrict__ wl, const float* __restrict__ bl,
    const float* __restrict__ wr, const float* __restrict__ br,
    const float* __restrict__ att, const float* __restrict__ bias,
    const unsigned char* __restrict__ adj,
    const float* __restrict__ skip,   // nullable
    float* __restrict__ out)
{
    extern __shared__ float smem[];
    float*    xlS    = smem + LDS_XL;             // f32 [128][64]
    float*    xrS    = smem + LDS_XR;             // f32 [128][64]
    _Float16* alphaS = (_Float16*)(smem + LDS_XR); // overlaps xrS exactly
    float*    attS   = smem + LDS_ATT;            // f32 [64]
    _Float16* xlTS   = (_Float16*)(smem + LDS_XLT); // f16 [64][128] (h-major)

    const int b   = blockIdx.x;
    const int tid = threadIdx.x;
    const int wave = tid >> 5;
    const int lane = tid & 31;

    // ---- Phase A: xl = x@wl + bl, xr = x@wr + br; xlT = f16 transpose -----
    for (int idx = tid; idx < NPTS * HID; idx += 256) {
        const int n = idx >> 6;
        const int h = idx & 63;
        float al = bl[h];
        float ar = br[h];
        const size_t xbase = ((size_t)b * NPTS + n) * xstride + xoff;
        for (int c = 0; c < cin; ++c) {
            const float xv = x[xbase + c];
            al += xv * wl[c * 64 + h];
            ar += xv * wr[c * 64 + h];
        }
        xlS[idx] = al;
        xrS[idx] = ar;
        xlTS[h * NPTS + n] = (_Float16)al;   // transposed f16 copy for WMMA B
    }
    if (tid < 64) attS[tid] = att[tid];
    __syncthreads();

    // ---- Phase B: masked-softmax attention rows (per wave: 16 rows) -------
    // alpha row i overwrites xr row i (only consumer is this wave, done first)
    for (int i = wave * 16; i < wave * 16 + 16; ++i) {
        float sv[4];
        int   aj[4];
        #pragma unroll
        for (int t = 0; t < 4; ++t) {
            const int j = lane + t * 32;
            aj[t] = adj[((size_t)b * NPTS + i) * NPTS + j];
            float s = 0.0f;
            #pragma unroll 8
            for (int h = 0; h < HID; ++h) {
                float v = xrS[i * HID + h] + xlS[j * HID + h];
                v = (v > 0.0f) ? v : 0.2f * v;   // leaky_relu(0.2)
                s += attS[h] * v;
            }
            sv[t] = s;
        }
        float m = -3.0e38f;
        #pragma unroll
        for (int t = 0; t < 4; ++t) if (aj[t]) m = fmaxf(m, sv[t]);
        #pragma unroll
        for (int off = 16; off >= 1; off >>= 1)
            m = fmaxf(m, __shfl_xor(m, off, 32));
        float e[4];
        float sum = 0.0f;
        #pragma unroll
        for (int t = 0; t < 4; ++t) {
            e[t] = aj[t] ? __expf(sv[t] - m) : 0.0f;
            sum += e[t];
        }
        #pragma unroll
        for (int off = 16; off >= 1; off >>= 1)
            sum += __shfl_xor(sum, off, 32);
        const float inv = 1.0f / sum;
        #pragma unroll
        for (int t = 0; t < 4; ++t)
            alphaS[i * NPTS + lane + t * 32] = (_Float16)(e[t] * inv);
    }
    __syncthreads();

    // ---- Phase C: out = alpha @ xl  (WMMA f16 in, f32 acc) ----------------
    // wave w owns i-tile w (rows 16w..16w+15); 4 h-tiles of 16; K loop 4x32.
    const int iT   = wave;
    const int lrow = lane & 15;
    const int lhi  = lane >> 4;
    const int arow = iT * 16 + lrow;

    // A fragments are h-tile invariant: load all 4 k-tiles once.
    // A (16x32 f16): element s -> K = 16*laneHalf + s; both
    // 8-half groups are contiguous in LDS -> two b128 loads per fragment.
    const _Float16* aRow = alphaS + arow * NPTS;
    v16h afr[4];
    #pragma unroll
    for (int kt = 0; kt < 4; ++kt) {
        const int k0 = kt * 32;
        const v8h alo = *(const v8h*)(aRow + k0 + (lhi << 4));
        const v8h ahi = *(const v8h*)(aRow + k0 + 8 + (lhi << 4));
        afr[kt] = __builtin_shufflevector(alo, ahi,
                    0, 1, 2, 3, 4, 5, 6, 7, 8, 9, 10, 11, 12, 13, 14, 15);
    }

    #pragma unroll
    for (int nT = 0; nT < 4; ++nT) {
        const int h0 = nT * 16;
        // B (32x16 f16): lane column fixed -> 16 consecutive halfs of xlT row
        const _Float16* bPtr = xlTS + (h0 + lrow) * NPTS + (lhi << 4);
        v8f acc = {};
        #pragma unroll
        for (int kt = 0; kt < 4; ++kt) {
            const v16h bfrag = *(const v16h*)(bPtr + kt * 32);
            acc = __builtin_amdgcn_wmma_f32_16x16x32_f16(
                false, afr[kt], false, bfrag, (short)0, acc, false, false);
        }
        // epilogue: + bias (+ skip) -> relu -> global
        #pragma unroll
        for (int r = 0; r < 8; ++r) {
            const int row = iT * 16 + (lhi << 3) + r;
            const int col = h0 + lrow;
            float v = acc[r] + bias[col];
            if (skip) v += skip[((size_t)b * NPTS + row) * HID + col];
            v = fmaxf(v, 0.0f);
            out[((size_t)b * NPTS + row) * HID + col] = v;
        }
    }
}

// ---------------------------------------------------------------------------
// Output heads: logits = x3 @ lab_w + lab_b ; values = x4 @ val_w + val_b
// ---------------------------------------------------------------------------
__global__ __launch_bounds__(256) void head_kernel(
    const float* __restrict__ x3, const float* __restrict__ x4,
    const float* __restrict__ lab_w, const float* __restrict__ lab_b,
    const float* __restrict__ val_w, const float* __restrict__ val_b,
    float* __restrict__ outf)
{
    const int row = blockIdx.x * 256 + threadIdx.x;
    if (row >= BATCH * NPTS) return;
    float l0 = lab_b[0], l1 = lab_b[1], l2 = lab_b[2], l3 = lab_b[3];
    float v  = val_b[0];
    #pragma unroll 8
    for (int k = 0; k < HID; ++k) {
        const float a = x3[(size_t)row * HID + k];
        l0 += a * lab_w[k * 4 + 0];
        l1 += a * lab_w[k * 4 + 1];
        l2 += a * lab_w[k * 4 + 2];
        l3 += a * lab_w[k * 4 + 3];
        v  += x4[(size_t)row * HID + k] * val_w[k];
    }
    outf[LOGITS_OFF + (size_t)row * 4 + 0] = l0;
    outf[LOGITS_OFF + (size_t)row * 4 + 1] = l1;
    outf[LOGITS_OFF + (size_t)row * 4 + 2] = l2;
    outf[LOGITS_OFF + (size_t)row * 4 + 3] = l3;
    outf[VALUES_OFF + row] = v;
}

// ---------------------------------------------------------------------------
extern "C" void kernel_launch(void* const* d_in, const int* in_sizes, int n_in,
                              void* d_out, int out_size, void* d_ws, size_t ws_size,
                              hipStream_t stream)
{
    const float* batch = (const float*)d_in[0];
    const float* ew1 = (const float*)d_in[1];
    const float* eb1 = (const float*)d_in[2];
    const float* ew2 = (const float*)d_in[3];
    const float* eb2 = (const float*)d_in[4];
    const float* ew3 = (const float*)d_in[5];
    const float* eb3 = (const float*)d_in[6];
    // g1..g4: wl, bl, wr, br, att, b at 7..12, 13..18, 19..24, 25..30
    const float* g[4][6];
    for (int l = 0; l < 4; ++l)
        for (int p = 0; p < 6; ++p)
            g[l][p] = (const float*)d_in[7 + l * 6 + p];
    const float* skip_w = (const float*)d_in[31];
    const float* skip_b = (const float*)d_in[32];
    const float* lab_w  = (const float*)d_in[33];
    const float* lab_b  = (const float*)d_in[34];
    const float* val_w  = (const float*)d_in[35];
    const float* val_b  = (const float*)d_in[36];

    float* outf = (float*)d_out;
    float* wsf  = (float*)d_ws;
    float* skip = wsf + WS_SKIP;
    float* x1   = wsf + WS_X1;
    float* x2   = wsf + WS_X2;
    float* x3   = wsf + WS_X3;
    float* x4   = wsf + WS_X4;
    unsigned char* adj = (unsigned char*)(wsf + WS_ADJF);

    encoder_kernel<<<BATCH * NPTS, 64, 0, stream>>>(
        batch, ew1, eb1, ew2, eb2, ew3, eb3, skip_w, skip_b, outf, skip);

    mask_kernel<<<BATCH, 256, 0, stream>>>(outf + LATENT_OFF, outf, adj);

    const size_t lds = LDS_TOTAL * sizeof(float);  // 82,176 B dynamic LDS
    // g1: x = latent[...,2:3] (stride 3, offset 2, cin 1)
    gat_kernel<<<BATCH, 256, lds, stream>>>(
        outf + LATENT_OFF, 3, 2, 1,
        g[0][0], g[0][1], g[0][2], g[0][3], g[0][4], g[0][5],
        adj, nullptr, x1);
    gat_kernel<<<BATCH, 256, lds, stream>>>(
        x1, 64, 0, 64,
        g[1][0], g[1][1], g[1][2], g[1][3], g[1][4], g[1][5],
        adj, nullptr, x2);
    gat_kernel<<<BATCH, 256, lds, stream>>>(
        x2, 64, 0, 64,
        g[2][0], g[2][1], g[2][2], g[2][3], g[2][4], g[2][5],
        adj, skip, x3);
    gat_kernel<<<BATCH, 256, lds, stream>>>(
        x2, 64, 0, 64,
        g[3][0], g[3][1], g[3][2], g[3][3], g[3][4], g[3][5],
        adj, skip, x4);

    head_kernel<<<(BATCH * NPTS + 255) / 256, 256, 0, stream>>>(
        x3, x4, lab_w, lab_b, val_w, val_b, outf);
}
